// EvoformerBlock_8435315769812
// MI455X (gfx1250) — matrix operand order repaired and checked by reading, not compiled, not run on hardware
//
#include <hip/hip_runtime.h>

typedef _Float16 half4  __attribute__((ext_vector_type(4)));
typedef _Float16 half8  __attribute__((ext_vector_type(8)));
typedef _Float16 half16 __attribute__((ext_vector_type(16)));
typedef float    float8 __attribute__((ext_vector_type(8)));

#define TILE 32
#define LDSS 40   // LDS row stride in halves: 80 bytes -> 16B-aligned half8 loads

// ---------------------------------------------------------------------------
// Generic strided, batched GEMM:  C = act(alpha * A x B + bias)  (or +=)
// Element addressing: X[b0*sX0 + b1*sX1 + b2*sX2 + m*rsX + k(or n)*csX]
// b0 comes from blockIdx.x / ntx, (b1,b2) from blockIdx.z (b1 = z/nb2).
// Template modes (AM for A, BM for B):
//   0 = guarded per-element loads (arbitrary dims/strides)
//   1 = aligned per-element loads (dims multiple of 32, arbitrary strides)
//   2 = vectorized float4 loads   (unit col stride, 16B-alignable)
// ---------------------------------------------------------------------------
struct GP {
  const float* A; const float* B; const float* bias; float* C;
  int M, N, K;
  long rsA, csA, rsB, csB, rsC, csC;
  long sA0, sB0, sC0;
  long sA1, sB1, sC1;
  long sA2, sB2, sC2;
  int ntx, nb2;
  float alpha;
  int act;    // 0 none, 1 relu, 2 sigmoid
  int accum;  // 0 store, 1 +=
};

template <int AM, int BM>
__global__ __launch_bounds__(128) void k_gemm(GP p)
{
  __shared__ __align__(16) _Float16 As[2][TILE][LDSS];   // As[buf][m][k]
  __shared__ __align__(16) _Float16 Bs[2][TILE][LDSS];   // Bs[buf][n][k] (transposed)

  const int bx = blockIdx.x % p.ntx;
  const int b0 = blockIdx.x / p.ntx;
  const int b1 = blockIdx.z / p.nb2;
  const int b2 = blockIdx.z % p.nb2;

  const float* Ab = p.A + b0 * p.sA0 + b1 * p.sA1 + b2 * p.sA2;
  const float* Bb = p.B + b0 * p.sB0 + b1 * p.sB1 + b2 * p.sB2;
  float*       Cb = p.C + b0 * p.sC0 + b1 * p.sC1 + b2 * p.sC2;

  const int m0 = blockIdx.y * TILE;
  const int n0 = bx * TILE;

  const int tid  = threadIdx.x;
  const int lane = tid & 31;
  const int wid  = tid >> 5;           // 4 waves: 2x2 arrangement of 16x16 tiles
  const int wy   = wid >> 1, wx = wid & 1;

  // ISA 7.12.2 fragment coordinates (wave32):
  const int ar    = (lane & 15) + wy * 16;     // A row (M)
  const int akoff = (lane < 16) ? 0 : 8;       // A: lanes16-31 hold K+8 / K+24
  const int bn    = (lane & 15) + wx * 16;     // B col (N)
  const int bkoff = (lane < 16) ? 0 : 8;       // B: lanes16-31 hold K+8 / K+24

  float8 acc = {0.f,0.f,0.f,0.f,0.f,0.f,0.f,0.f};

  auto loadTile = [&](int buf, int k0) {
    if (AM == 2) {                                  // float4 (b128) loads, no guards
      #pragma unroll
      for (int t = 0; t < 2; ++t) {
        int i = tid + t * 128;                      // 256 float4s = 32 rows x 8
        int r = i >> 3, q = (i & 7) * 4;
        float4 f = *((const float4*)(Ab + (long)(m0 + r) * p.rsA + k0 + q));
        half4 h = { (_Float16)f.x, (_Float16)f.y, (_Float16)f.z, (_Float16)f.w };
        *(half4*)&As[buf][r][q] = h;                // 8-byte LDS store
      }
    } else {
      #pragma unroll
      for (int t = 0; t < 8; ++t) {
        int i = tid + t * 128;
        int r = i >> 5, c = i & 31;
        int m = m0 + r, k = k0 + c;
        float av;
        if (AM == 1) av = Ab[(long)m * p.rsA + (long)k * p.csA];
        else av = (m < p.M && k < p.K) ? Ab[(long)m * p.rsA + (long)k * p.csA] : 0.f;
        As[buf][r][c] = (_Float16)av;
      }
    }
    if (BM == 2) {
      #pragma unroll
      for (int t = 0; t < 2; ++t) {
        int i = tid + t * 128;                      // row k = r, cols n0+q..n0+q+3
        int r = i >> 3, q = (i & 7) * 4;
        float4 f = *((const float4*)(Bb + (long)(k0 + r) * p.rsB + n0 + q));
        Bs[buf][q + 0][r] = (_Float16)f.x;          // transposed scatter into LDS
        Bs[buf][q + 1][r] = (_Float16)f.y;
        Bs[buf][q + 2][r] = (_Float16)f.z;
        Bs[buf][q + 3][r] = (_Float16)f.w;
      }
    } else {
      #pragma unroll
      for (int t = 0; t < 8; ++t) {
        int i = tid + t * 128;
        int r = i >> 5, c = i & 31;
        int kk = k0 + r, n = n0 + c;
        float bv;
        if (BM == 1) bv = Bb[(long)kk * p.rsB + (long)n * p.csB];
        else bv = (kk < p.K && n < p.N) ? Bb[(long)kk * p.rsB + (long)n * p.csB] : 0.f;
        Bs[buf][c][r] = (_Float16)bv;               // transposed store: Bs[n][k]
      }
    }
  };

  int buf = 0;
  loadTile(0, 0);
  for (int k0 = 0; k0 < p.K; k0 += TILE) {
    __syncthreads();                    // tile[buf] ready; tile[buf^1] free
    int kn = k0 + TILE;
    if (kn < p.K) loadTile(buf ^ 1, kn);   // overlap next-tile loads with WMMA

    half8 ha0 = *(const half8*)&As[buf][ar][akoff];        // K = koff..koff+7
    half8 ha1 = *(const half8*)&As[buf][ar][akoff + 16];   // K = 16+koff..23+koff
    half8 hb0 = *(const half8*)&Bs[buf][bn][bkoff];
    half8 hb1 = *(const half8*)&Bs[buf][bn][bkoff + 16];
    half16 af = __builtin_shufflevector(ha0, ha1, 0,1,2,3,4,5,6,7,8,9,10,11,12,13,14,15);
    half16 bf = __builtin_shufflevector(hb0, hb1, 0,1,2,3,4,5,6,7,8,9,10,11,12,13,14,15);
#if defined(__HIP_DEVICE_COMPILE__) && defined(__gfx1250__)
    acc = __builtin_amdgcn_wmma_f32_16x16x32_f16(false, af, false, bf,
                                                 (short)0, acc, false, false);
#endif
    buf ^= 1;
  }

  // C/D layout: VGPR r -> M = r (lanes 0-15) or M = 8+r (lanes 16-31), N = lane&15
  const int nn = n0 + wx * 16 + (lane & 15);
  const int mb = m0 + wy * 16 + ((lane < 16) ? 0 : 8);
  if (nn < p.N) {
    float bv = p.bias ? p.bias[nn] : 0.f;
    #pragma unroll
    for (int r = 0; r < 8; ++r) {
      int m = mb + r;
      if (m < p.M) {
        float v = p.alpha * acc[r] + bv;
        if (p.act == 1)      v = v > 0.f ? v : 0.f;
        else if (p.act == 2) v = 1.f / (1.f + __expf(-v));
        long addr = (long)m * p.rsC + (long)nn * p.csC;
        if (p.accum) Cb[addr] += v; else Cb[addr] = v;
      }
    }
  }
}

// ---------------------------------------------------------------------------
// Row-wise kernels (one wave32 per row)
// ---------------------------------------------------------------------------
__global__ __launch_bounds__(256)
void k_layernorm(const float* __restrict__ x, const float* __restrict__ g,
                 const float* __restrict__ b, float* __restrict__ y,
                 int rows, int C)
{
  int row = blockIdx.x * 8 + (threadIdx.x >> 5);
  if (row >= rows) return;
  int lane = threadIdx.x & 31;
  const float* xr = x + (size_t)row * C;
  float* yr = y + (size_t)row * C;
  float s = 0.f, ss = 0.f;
  for (int j = lane; j < C; j += 32) { float t = xr[j]; s += t; ss += t * t; }
  for (int o = 16; o > 0; o >>= 1) { s += __shfl_xor(s, o, 32); ss += __shfl_xor(ss, o, 32); }
  float mu  = s / (float)C;
  float var = ss / (float)C - mu * mu;
  float rstd = rsqrtf(var + 1e-5f);
  for (int j = lane; j < C; j += 32) yr[j] = (xr[j] - mu) * rstd * g[j] + b[j];
}

// softmax over last dim with mask bias (per b1 row) and optional [h,q,k] bias
__global__ __launch_bounds__(256)
void k_softmax(float* __restrict__ s, const float* __restrict__ mask,
               const float* __restrict__ pb, int nrows, int L, int H, int Q)
{
  int row = blockIdx.x * 8 + (threadIdx.x >> 5);
  if (row >= nrows) return;
  int lane = threadIdx.x & 31;
  int b1 = row / (H * Q);
  int hq = row % (H * Q);
  float* r = s + (size_t)row * L;
  const float* mr = mask + (size_t)b1 * L;
  const float* pr = pb ? pb + (size_t)hq * L : nullptr;
  float v[8];
  int cnt = 0;
  float mx = -3.0e38f;
  for (int j = lane; j < L; j += 32) {
    float t = r[j] + 1e9f * (mr[j] - 1.0f);
    if (pr) t += pr[j];
    v[cnt++] = t;
    mx = fmaxf(mx, t);
  }
  for (int o = 16; o > 0; o >>= 1) mx = fmaxf(mx, __shfl_xor(mx, o, 32));
  float sum = 0.f; cnt = 0;
  for (int j = lane; j < L; j += 32) { float e = __expf(v[cnt] - mx); v[cnt++] = e; sum += e; }
  for (int o = 16; o > 0; o >>= 1) sum += __shfl_xor(sum, o, 32);
  float inv = 1.0f / sum;
  cnt = 0;
  for (int j = lane; j < L; j += 32) r[j] = v[cnt++] * inv;
}

// ---------------------------------------------------------------------------
// Elementwise helpers
// ---------------------------------------------------------------------------
__global__ void k_copy(float* d, const float* s, size_t n)
{ size_t i = (size_t)blockIdx.x * 256 + threadIdx.x; if (i < n) d[i] = s[i]; }

__global__ void k_mul(float* d, const float* s, size_t n)
{ size_t i = (size_t)blockIdx.x * 256 + threadIdx.x; if (i < n) d[i] *= s[i]; }

__global__ void k_add_mul(float* d, const float* a, const float* b, size_t n)
{ size_t i = (size_t)blockIdx.x * 256 + threadIdx.x; if (i < n) d[i] += a[i] * b[i]; }

__global__ void k_mask_rows(float* x, const float* mask, size_t n, int C)
{ size_t i = (size_t)blockIdx.x * 256 + threadIdx.x; if (i < n) x[i] *= mask[i / C]; }

__global__ void k_add_masked(float* d, const float* s, const float* mask, size_t n, int C)
{ size_t i = (size_t)blockIdx.x * 256 + threadIdx.x; if (i < n) d[i] += s[i] * mask[i / C]; }

__global__ void k_mul3_mask(float* d, const float* mask, const float* g, const float* p,
                            size_t n, int C)
{ size_t i = (size_t)blockIdx.x * 256 + threadIdx.x; if (i < n) d[i] = mask[i / C] * g[i] * p[i]; }

__global__ void k_add_div(float* d, const float* s, const float* nrm, size_t n, int C)
{ size_t i = (size_t)blockIdx.x * 256 + threadIdx.x; if (i < n) d[i] += s[i] / (1e-3f + nrm[i / C]); }

__global__ void k_transpose(float* dst, const float* src, int D0, int D1, int C)
{
  size_t n = (size_t)D0 * D1 * C;
  size_t i = (size_t)blockIdx.x * 256 + threadIdx.x;
  if (i >= n) return;
  int c = (int)(i % C);
  size_t t = i / C;
  int d1 = (int)(t % D1);
  int d0 = (int)(t / D1);
  dst[((size_t)d1 * D0 + d0) * C + c] = src[i];
}

__global__ void k_opm_norm(float* nrm, const float* mask, int S_, int R_)
{
  int idx = blockIdx.x * 256 + threadIdx.x;
  if (idx >= R_ * R_) return;
  int i = idx / R_, j = idx % R_;
  float s = 0.f;
  for (int t = 0; t < S_; ++t) s += mask[t * R_ + i] * mask[t * R_ + j];
  nrm[idx] = s;
}

// ---------------------------------------------------------------------------
// Host-side helpers
// ---------------------------------------------------------------------------
static inline unsigned blk(size_t n) { return (unsigned)((n + 255) / 256); }
static inline int divup(long a, long b) { return (int)((a + b - 1) / b); }

static void gemm(hipStream_t st, const float* A, const float* B, const float* bias, float* C,
                 int M, int N, int K,
                 long rsA, long csA, long rsB, long csB, long rsC, long csC,
                 float alpha, int act, int accum,
                 int b0n = 1, long sA0 = 0, long sB0 = 0, long sC0 = 0,
                 int b1n = 1, long sA1 = 0, long sB1 = 0, long sC1 = 0,
                 int b2n = 1, long sA2 = 0, long sB2 = 0, long sC2 = 0)
{
  GP p;
  p.A = A; p.B = B; p.bias = bias; p.C = C; p.M = M; p.N = N; p.K = K;
  p.rsA = rsA; p.csA = csA; p.rsB = rsB; p.csB = csB; p.rsC = rsC; p.csC = csC;
  p.sA0 = sA0; p.sB0 = sB0; p.sC0 = sC0;
  p.sA1 = sA1; p.sB1 = sB1; p.sC1 = sC1;
  p.sA2 = sA2; p.sB2 = sB2; p.sC2 = sC2;
  p.ntx = divup(N, 32); p.nb2 = b2n;
  p.alpha = alpha; p.act = act; p.accum = accum;

  bool alA = (M % 32 == 0) && (K % 32 == 0);
  bool alB = (N % 32 == 0) && (K % 32 == 0);
  int am = (alA && csA == 1 && (rsA % 4 == 0) &&
            (sA0 % 4 == 0) && (sA1 % 4 == 0) && (sA2 % 4 == 0)) ? 2 : (alA ? 1 : 0);
  int bm = (alB && csB == 1 && (rsB % 4 == 0) &&
            (sB0 % 4 == 0) && (sB1 % 4 == 0) && (sB2 % 4 == 0)) ? 2 : (alB ? 1 : 0);

  dim3 grid((unsigned)(p.ntx * b0n), (unsigned)divup(M, 32), (unsigned)(b1n * b2n));
  switch (am * 3 + bm) {
    case 0: k_gemm<0,0><<<grid, 128, 0, st>>>(p); break;
    case 1: k_gemm<0,1><<<grid, 128, 0, st>>>(p); break;
    case 2: k_gemm<0,2><<<grid, 128, 0, st>>>(p); break;
    case 3: k_gemm<1,0><<<grid, 128, 0, st>>>(p); break;
    case 4: k_gemm<1,1><<<grid, 128, 0, st>>>(p); break;
    case 5: k_gemm<1,2><<<grid, 128, 0, st>>>(p); break;
    case 6: k_gemm<2,0><<<grid, 128, 0, st>>>(p); break;
    case 7: k_gemm<2,1><<<grid, 128, 0, st>>>(p); break;
    default: k_gemm<2,2><<<grid, 128, 0, st>>>(p); break;
  }
}

static void lnorm(hipStream_t st, const float* x, const float* g, const float* b,
                  float* y, int rows, int C)
{ k_layernorm<<<divup(rows, 8), 256, 0, st>>>(x, g, b, y, rows, C); }

static void softmaxl(hipStream_t st, float* s, const float* mask, const float* pb,
                     int nrows, int L, int H, int Q)
{ k_softmax<<<divup(nrows, 8), 256, 0, st>>>(s, mask, pb, nrows, L, H, Q); }

// ---------------------------------------------------------------------------
extern "C" void kernel_launch(void* const* d_in, const int* in_sizes, int n_in,
                              void* d_out, int out_size, void* d_ws, size_t ws_size,
                              hipStream_t stream)
{
  (void)in_sizes; (void)n_in; (void)out_size; (void)ws_size;

  const int S = 128, R = 256, CM = 256, CZ = 128;
  const int HM = 8, HP = 4, CH = 32, DM = 256, DP = 128, HO = 32;
  const float ISQ = 0.17677669529663687f;   // 1/sqrt(32)

  // params flattened as a jax pytree: dict keys sorted alphabetically per level
  const int COL = 4;   // bg bo ln_b ln_g wg wk wo wq wv
  const int MT  = 13;  // b1 b2 ln_b ln_g w1 w2
  const int OPM = 19;  // ba bb bo ln_b ln_g wa wb wo
  const int PT  = 27;  // b1 b2 ln_b ln_g w1 w2
  const int ROW = 33;  // bg bo ln_m_b ln_m_g ln_z_b ln_z_g wg wk wo wq wv wz
  const int TAE = 45;  // bg bo ln_b ln_g wg wk wo wq wv wz
  const int TAS = 55;
  const int TMI = 65;  // bag bap bbg bbp bg bp ln_b ln_g ln_o_b ln_o_g wag wap wbg wbp wg wp
  const int TMO = 81;

  auto PF = [&](int i) { return (const float*)d_in[i]; };

  const float* m_in      = PF(0);
  const float* z_in      = PF(1);
  const float* msa_mask  = PF(2);
  const float* pair_mask = PF(3);

  // workspace layout (floats). S*R*CM == R*R*CZ == 8388608
  const size_t NB = 8388608;
  float* W   = (float*)d_ws;
  float* xln = W;
  float* zln = W + NB;
  float* qb  = W + 2 * NB;
  float* kb  = W + 3 * NB;
  float* vb  = W + 4 * NB;
  float* gb  = W + 5 * NB;
  float* ob  = W + 6 * NB;
  float* tA  = W + 7 * NB;
  float* tB  = W + 8 * NB;
  float* tT  = W + 9 * NB;
  float* BIG = W + 10 * NB;          // 67108864 floats (scores / OPM outer / hidden)
  float* pbb = BIG + 67108864;       // 524288  (attention pair/tri bias [h,q,k])
  float* nrm = pbb + 524288;         // 65536   (OPM normalization)

  float* out_m = (float*)d_out;      // running m state (residual accumulated)
  float* out_z = out_m + NB;         // running z state

  k_copy<<<blk(NB), 256, 0, stream>>>(out_m, m_in, NB);
  k_copy<<<blk(NB), 256, 0, stream>>>(out_z, z_in, NB);

  // ================= 1) MSA row attention (with pair bias) =================
  lnorm(stream, out_m, PF(ROW + 3), PF(ROW + 2), xln, S * R, CM);
  lnorm(stream, out_z, PF(ROW + 5), PF(ROW + 4), zln, R * R, CZ);
  gemm(stream, zln, PF(ROW + 11), nullptr, pbb, R * R, HM, CZ,
       CZ, 1, HM, 1, 1, (long)R * R, 1.f, 0, 0);
  gemm(stream, xln, PF(ROW + 9),  nullptr, qb, S * R, DM, CM, CM, 1, DM, 1, DM, 1, 1.f, 0, 0);
  gemm(stream, xln, PF(ROW + 7),  nullptr, kb, S * R, DM, CM, CM, 1, DM, 1, DM, 1, 1.f, 0, 0);
  gemm(stream, xln, PF(ROW + 10), nullptr, vb, S * R, DM, CM, CM, 1, DM, 1, DM, 1, 1.f, 0, 0);
  gemm(stream, xln, PF(ROW + 6), PF(ROW + 0), gb, S * R, DM, CM, CM, 1, DM, 1, DM, 1, 1.f, 2, 0);
  gemm(stream, qb, kb, nullptr, BIG, R, R, CH,            // scores[s,h,q,k]
       DM, 1, 1, DM, R, 1, ISQ, 0, 0,
       1, 0, 0, 0,
       S, (long)R * DM, (long)R * DM, (long)HM * R * R,
       HM, CH, CH, (long)R * R);
  softmaxl(stream, BIG, msa_mask, pbb, S * HM * R, R, HM, R);
  gemm(stream, BIG, vb, nullptr, ob, R, CH, R,            // o = A @ V
       R, 1, DM, 1, DM, 1, 1.f, 0, 0,
       1, 0, 0, 0,
       S, (long)HM * R * R, (long)R * DM, (long)R * DM,
       HM, (long)R * R, CH, CH);
  k_mul<<<blk(NB), 256, 0, stream>>>(ob, gb, NB);
  gemm(stream, ob, PF(ROW + 8), PF(ROW + 1), out_m, S * R, CM, DM,
       DM, 1, CM, 1, CM, 1, 1.f, 0, 1);

  // ================= 2) MSA column attention =================
  k_transpose<<<blk(NB), 256, 0, stream>>>(tT, out_m, S, R, CM);
  k_transpose<<<blk((size_t)S * R), 256, 0, stream>>>(tA, msa_mask, S, R, 1);
  lnorm(stream, tT, PF(COL + 3), PF(COL + 2), xln, R * S, CM);
  gemm(stream, xln, PF(COL + 7), nullptr, qb, R * S, DM, CM, CM, 1, DM, 1, DM, 1, 1.f, 0, 0);
  gemm(stream, xln, PF(COL + 5), nullptr, kb, R * S, DM, CM, CM, 1, DM, 1, DM, 1, 1.f, 0, 0);
  gemm(stream, xln, PF(COL + 8), nullptr, vb, R * S, DM, CM, CM, 1, DM, 1, DM, 1, 1.f, 0, 0);
  gemm(stream, xln, PF(COL + 4), PF(COL + 0), gb, R * S, DM, CM, CM, 1, DM, 1, DM, 1, 1.f, 2, 0);
  gemm(stream, qb, kb, nullptr, BIG, S, S, CH,
       DM, 1, 1, DM, S, 1, ISQ, 0, 0,
       1, 0, 0, 0,
       R, (long)S * DM, (long)S * DM, (long)HM * S * S,
       HM, CH, CH, (long)S * S);
  softmaxl(stream, BIG, tA, nullptr, R * HM * S, S, HM, S);
  gemm(stream, BIG, vb, nullptr, ob, S, CH, S,
       S, 1, DM, 1, DM, 1, 1.f, 0, 0,
       1, 0, 0, 0,
       R, (long)HM * S * S, (long)S * DM, (long)S * DM,
       HM, (long)S * S, CH, CH);
  k_mul<<<blk(NB), 256, 0, stream>>>(ob, gb, NB);
  gemm(stream, ob, PF(COL + 6), PF(COL + 1), out_m, S, CM, DM,   // transposed accum
       DM, 1, CM, 1, (long)R * CM, 1, 1.f, 0, 1,
       1, 0, 0, 0,
       R, (long)S * DM, 0, (long)CM,
       1, 0, 0, 0);

  // ================= transitions helper =================
  auto trans_mod = [&](int base, float* out, const float* mask, int rows, int C, int NH) {
    lnorm(stream, out, PF(base + 3), PF(base + 2), xln, rows, C);
    gemm(stream, xln, PF(base + 4), PF(base + 0), BIG, rows, NH, C,
         C, 1, NH, 1, NH, 1, 1.f, 1, 0);
    gemm(stream, BIG, PF(base + 5), PF(base + 1), ob, rows, C, NH,
         NH, 1, C, 1, C, 1, 1.f, 0, 0);
    k_add_masked<<<blk((size_t)rows * C), 256, 0, stream>>>(out, ob, mask, (size_t)rows * C, C);
  };

  // ================= 3) MSA transition =================
  trans_mod(MT, out_m, msa_mask, S * R, CM, 4 * CM);

  // ================= 4) Outer product mean =================
  lnorm(stream, out_m, PF(OPM + 4), PF(OPM + 3), xln, S * R, CM);
  gemm(stream, xln, PF(OPM + 5), PF(OPM + 0), tA, S * R, HO, CM, CM, 1, HO, 1, HO, 1, 1.f, 0, 0);
  k_mask_rows<<<blk((size_t)S * R * HO), 256, 0, stream>>>(tA, msa_mask, (size_t)S * R * HO, HO);
  gemm(stream, xln, PF(OPM + 6), PF(OPM + 1), tB, S * R, HO, CM, CM, 1, HO, 1, HO, 1, 1.f, 0, 0);
  k_mask_rows<<<blk((size_t)S * R * HO), 256, 0, stream>>>(tB, msa_mask, (size_t)S * R * HO, HO);
  gemm(stream, tA, tB, nullptr, BIG, HO, HO, S,   // outer[i,j,c,d]; batch i in z, j in x
       1, (long)R * HO, (long)R * HO, 1, HO, 1, 1.f, 0, 0,
       R, 0, HO, (long)HO * HO,
       R, HO, 0, (long)R * HO * HO,
       1, 0, 0, 0);
  gemm(stream, BIG, PF(OPM + 7), PF(OPM + 2), ob, R * R, CZ, HO * HO,
       (long)HO * HO, 1, CZ, 1, CZ, 1, 1.f, 0, 0);
  k_opm_norm<<<blk((size_t)R * R), 256, 0, stream>>>(nrm, msa_mask, S, R);
  k_add_div<<<blk((size_t)R * R * CZ), 256, 0, stream>>>(out_z, ob, nrm, (size_t)R * R * CZ, CZ);

  // ================= triangle multiplication helper =================
  auto tri_mul_mod = [&](int base, int outgoing) {
    lnorm(stream, out_z, PF(base + 7), PF(base + 6), zln, R * R, CZ);
    gemm(stream, zln, PF(base + 10), PF(base + 0), tA, R * R, CZ, CZ, CZ, 1, CZ, 1, CZ, 1, 1.f, 2, 0);
    gemm(stream, zln, PF(base + 11), PF(base + 1), tB, R * R, CZ, CZ, CZ, 1, CZ, 1, CZ, 1, 1.f, 0, 0);
    k_mul3_mask<<<blk((size_t)R * R * CZ), 256, 0, stream>>>(qb, pair_mask, tA, tB, (size_t)R * R * CZ, CZ);
    gemm(stream, zln, PF(base + 12), PF(base + 2), tA, R * R, CZ, CZ, CZ, 1, CZ, 1, CZ, 1, 1.f, 2, 0);
    gemm(stream, zln, PF(base + 13), PF(base + 3), tB, R * R, CZ, CZ, CZ, 1, CZ, 1, CZ, 1, 1.f, 0, 0);
    k_mul3_mask<<<blk((size_t)R * R * CZ), 256, 0, stream>>>(kb, pair_mask, tA, tB, (size_t)R * R * CZ, CZ);
    if (outgoing)   // x[i,j,c] = sum_k a[i,k,c] b[j,k,c], batch over c
      gemm(stream, qb, kb, nullptr, vb, R, R, R,
           (long)R * CZ, CZ, CZ, (long)R * CZ, (long)R * CZ, CZ, 1.f, 0, 0,
           1, 0, 0, 0, CZ, 1, 1, 1, 1, 0, 0, 0);
    else            // x[i,j,c] = sum_k a[k,i,c] b[k,j,c]
      gemm(stream, qb, kb, nullptr, vb, R, R, R,
           CZ, (long)R * CZ, (long)R * CZ, CZ, (long)R * CZ, CZ, 1.f, 0, 0,
           1, 0, 0, 0, CZ, 1, 1, 1, 1, 0, 0, 0);
    lnorm(stream, vb, PF(base + 9), PF(base + 8), ob, R * R, CZ);
    gemm(stream, ob, PF(base + 15), PF(base + 5), tA, R * R, CZ, CZ, CZ, 1, CZ, 1, CZ, 1, 1.f, 0, 0);
    gemm(stream, zln, PF(base + 14), PF(base + 4), tB, R * R, CZ, CZ, CZ, 1, CZ, 1, CZ, 1, 1.f, 2, 0);
    k_add_mul<<<blk((size_t)R * R * CZ), 256, 0, stream>>>(out_z, tA, tB, (size_t)R * R * CZ);
  };

  // ================= 5,6) triangle mult outgoing / incoming =================
  tri_mul_mod(TMO, 1);
  tri_mul_mod(TMI, 0);

  // ================= triangle attention helper =================
  auto tri_attn_mod = [&](int base, int starting) {
    const float* src; const float* mk;
    if (starting) { src = out_z; mk = pair_mask; }
    else {
      k_transpose<<<blk((size_t)R * R * CZ), 256, 0, stream>>>(tT, out_z, R, R, CZ);
      k_transpose<<<blk((size_t)R * R), 256, 0, stream>>>(tB, pair_mask, R, R, 1);
      src = tT; mk = tB;
    }
    lnorm(stream, src, PF(base + 3), PF(base + 2), zln, R * R, CZ);
    gemm(stream, zln, PF(base + 9), nullptr, pbb, R * R, HP, CZ,
         CZ, 1, HP, 1, 1, (long)R * R, 1.f, 0, 0);
    gemm(stream, zln, PF(base + 7), nullptr, qb, R * R, DP, CZ, CZ, 1, DP, 1, DP, 1, 1.f, 0, 0);
    gemm(stream, zln, PF(base + 5), nullptr, kb, R * R, DP, CZ, CZ, 1, DP, 1, DP, 1, 1.f, 0, 0);
    gemm(stream, zln, PF(base + 8), nullptr, vb, R * R, DP, CZ, CZ, 1, DP, 1, DP, 1, 1.f, 0, 0);
    gemm(stream, zln, PF(base + 4), PF(base + 0), gb, R * R, DP, CZ, CZ, 1, DP, 1, DP, 1, 1.f, 2, 0);
    gemm(stream, qb, kb, nullptr, BIG, R, R, CH,
         DP, 1, 1, DP, R, 1, ISQ, 0, 0,
         1, 0, 0, 0,
         R, (long)R * DP, (long)R * DP, (long)HP * R * R,
         HP, CH, CH, (long)R * R);
    softmaxl(stream, BIG, mk, pbb, R * HP * R, R, HP, R);
    gemm(stream, BIG, vb, nullptr, ob, R, CH, R,
         R, 1, DP, 1, DP, 1, 1.f, 0, 0,
         1, 0, 0, 0,
         R, (long)HP * R * R, (long)R * DP, (long)R * DP,
         HP, (long)R * R, CH, CH);
    k_mul<<<blk((size_t)R * R * DP), 256, 0, stream>>>(ob, gb, (size_t)R * R * DP);
    if (starting)
      gemm(stream, ob, PF(base + 6), PF(base + 1), out_z, R * R, CZ, DP,
           DP, 1, CZ, 1, CZ, 1, 1.f, 0, 1);
    else  // accumulate transposed back into out_z
      gemm(stream, ob, PF(base + 6), PF(base + 1), out_z, R, CZ, DP,
           DP, 1, CZ, 1, (long)R * CZ, 1, 1.f, 0, 1,
           1, 0, 0, 0,
           R, (long)R * DP, 0, (long)CZ,
           1, 0, 0, 0);
  };

  // ================= 7,8) triangle attention starting / ending =================
  tri_attn_mod(TAS, 1);
  tri_attn_mod(TAE, 0);

  // ================= 9) pair transition =================
  trans_mod(PT, out_z, pair_mask, R * R, CZ, 4 * CZ);
}
